// EqualizedModConv2d_20590073217707
// MI455X (gfx1250) — matrix operand order repaired and checked by reading, not compiled, not run on hardware
//
#include <hip/hip_runtime.h>
#include <hip/hip_bf16.h>

typedef __attribute__((ext_vector_type(16))) __bf16 v16bf;
typedef __attribute__((ext_vector_type(8)))  float  v8f;
typedef __attribute__((ext_vector_type(4)))  int    i4;

#define C_IN  512
#define C_OUT 512
#define NB    8
#define DLAT  512
#define HP    34      // padded H/W
#define NPIX  1024    // 32*32
#define LDSP  40      // padded ci-stride in LDS (bank-conflict-free)

// ---------------- style: s[b][ci] = sqrt2 * lrelu(y @ (mw * dlat^-0.5)^T + bias)
__global__ void style_k(const float* __restrict__ y, const float* __restrict__ mw,
                        const float* __restrict__ bias, float* __restrict__ s) {
  int idx = blockIdx.x * blockDim.x + threadIdx.x;   // b*512+ci
  int b = idx >> 9, ci = idx & 511;
  const float* yb = y + b * DLAT;
  const float* mr = mw + (size_t)ci * DLAT;
  float acc = 0.f;
  for (int d = 0; d < DLAT; ++d) acc += yb[d] * mr[d];
  acc = acc * 0.04419417382415922f + bias[ci];       // dlat^-0.5
  acc = (acc > 0.f ? acc : 0.2f * acc) * 1.4142135623730951f;
  s[idx] = acc;
}

// ---------------- wsq[co][ci] = sum_kk weight^2
__global__ void wsq_k(const float* __restrict__ w, float* __restrict__ wsq) {
  int idx = blockIdx.x * blockDim.x + threadIdx.x;   // co*512+ci
  const float* p = w + (size_t)idx * 9;
  float a = 0.f;
  for (int t = 0; t < 9; ++t) a += p[t] * p[t];
  wsq[idx] = a;
}

// ---------------- d[b][co] = rsqrt(sum_ci s^2 * wsq + eps)
__global__ void demod_k(const float* __restrict__ s, const float* __restrict__ wsq,
                        float* __restrict__ dm) {
  int idx = blockIdx.x * blockDim.x + threadIdx.x;   // b*512+co
  int b = idx >> 9, co = idx & 511;
  const float* sb = s + b * C_IN;
  const float* wr = wsq + (size_t)co * C_IN;
  float acc = 1e-8f;
  for (int ci = 0; ci < C_IN; ++ci) { float sv = sb[ci]; acc += sv * sv * wr[ci]; }
  dm[idx] = rsqrtf(acc);
}

// ---------------- pack x*s -> padded NHWC bf16 [B][34][34][512]
__global__ void packx_k(const float* __restrict__ x, const float* __restrict__ s,
                        __bf16* __restrict__ xs) {
  int o = blockIdx.x * blockDim.x + threadIdx.x;     // ((b*34+r)*34+c)*512+ci
  int ci = o & 511;
  int pix = o >> 9;
  int c = pix % HP;
  int rb = pix / HP;
  int r = rb % HP;
  int b = rb / HP;
  float v = 0.f;
  if (r >= 1 && r <= 32 && c >= 1 && c <= 32) {
    v = x[((size_t)(b * C_IN + ci)) * NPIX + (r - 1) * 32 + (c - 1)] * s[b * C_IN + ci];
  }
  xs[o] = (__bf16)v;
}

// ---------------- pack weight -> [tap][co][ci] bf16
__global__ void packw_k(const float* __restrict__ w, __bf16* __restrict__ wp) {
  int o = blockIdx.x * blockDim.x + threadIdx.x;     // (tap*512+co)*512+ci
  int ci = o & 511;
  int t2 = o >> 9;
  int co = t2 & 511;
  int tap = t2 >> 9;
  wp[o] = (__bf16)w[((size_t)(co * C_IN + ci)) * 9 + tap];
}

// ---------------- main: implicit-GEMM 3x3 conv with bf16 WMMA
// grid (pxBlk=8, coBlk=8, b=8), block 256 (8 waves -> 2(M) x 4(N))
__global__ __launch_bounds__(256) void modconv_wmma(
    const __bf16* __restrict__ Xs,   // [B][34][34][512]
    const __bf16* __restrict__ Wp,   // [9][512][512]
    const float*  __restrict__ dmod, // [B][512]
    float*        __restrict__ out)  // [B][512][1024]
{
  __shared__ __bf16 lds[2][6 * HP * LDSP];           // double-buffered halo tiles

  const int tid   = threadIdx.x;
  const int b     = blockIdx.z, coBlk = blockIdx.y, pxBlk = blockIdx.x;
  const int wave  = tid >> 5, lane = tid & 31;
  const int waveM = wave >> 2, waveN = wave & 3;
  const int l15   = lane & 15, half = lane >> 4;
  const int r0    = pxBlk * 4;                       // first padded row of halo

  const __bf16* xbase = Xs + ((size_t)b * HP + r0) * HP * C_IN;
  const unsigned ldsbase = (unsigned)(size_t)(&lds[0][0]);

  // async-copy one ci-chunk's halo region (6 rows x 34 cols x 32 ci) into LDS
  auto issue = [&](int ci0, int buf) {
    unsigned lb = ldsbase + (unsigned)buf * (6 * HP * LDSP * 2);
    const __bf16* gb = xbase + ci0;
    for (int u = tid; u < 6 * HP * 4; u += 256) {    // 816 x 16B units
      int q = u & 3, pix = u >> 2;                   // pix = row*34+col
      unsigned long long ga = (unsigned long long)(size_t)(gb + (size_t)pix * C_IN + q * 8);
      unsigned la = lb + (unsigned)(pix * (LDSP * 2) + q * 16);
      asm volatile("global_load_async_to_lds_b128 %0, %1, off"
                   :: "v"(la), "v"(ga) : "memory");
    }
  };

  v8f acc[2][2] = {};

  issue(0, 0);
  for (int k = 0; k < 16; ++k) {
    asm volatile("s_wait_asynccnt 0x0" ::: "memory");
    __syncthreads();
    if (k + 1 < 16) issue((k + 1) * 32, (k + 1) & 1);

    const __bf16* lbuf = &lds[k & 1][0];
    const int ci0 = k * 32;

    for (int tap = 0; tap < 9; ++tap) {
      const int kh = tap / 3, kw = tap - kh * 3;

      union { v16bf v; i4 q[2]; } a0, a1, b0, b1;
      // A fragments (weights) straight from L2-resident Wp: K={0..7,16..23}+8*half
      {
        const __bf16* ap0 =
            Wp + ((size_t)tap * C_OUT + (coBlk * 64 + waveM * 32 + l15)) * C_IN + ci0;
        a0.q[0] = *(const i4*)(ap0 + 8 * half);
        a0.q[1] = *(const i4*)(ap0 + 16 + 8 * half);
        const __bf16* ap1 = ap0 + (size_t)16 * C_IN;
        a1.q[0] = *(const i4*)(ap1 + 8 * half);
        a1.q[1] = *(const i4*)(ap1 + 16 + 8 * half);
      }
      // B fragments (activations) from LDS halo: K={0..7,16..23}+8*half, N = pixel
      {
        int pix0 = waveN * 32 + l15;
        int ph0 = (pix0 >> 5) + kh, pw0 = (pix0 & 31) + kw;
        const __bf16* bp0 = lbuf + (ph0 * HP + pw0) * LDSP;
        b0.q[0] = *(const i4*)(bp0 + 8 * half);
        b0.q[1] = *(const i4*)(bp0 + 16 + 8 * half);
        int pix1 = pix0 + 16;
        int ph1 = (pix1 >> 5) + kh, pw1 = (pix1 & 31) + kw;
        const __bf16* bp1 = lbuf + (ph1 * HP + pw1) * LDSP;
        b1.q[0] = *(const i4*)(bp1 + 8 * half);
        b1.q[1] = *(const i4*)(bp1 + 16 + 8 * half);
      }

      acc[0][0] = __builtin_amdgcn_wmma_f32_16x16x32_bf16(false, a0.v, false, b0.v,
                                                          (short)0, acc[0][0], false, false);
      acc[0][1] = __builtin_amdgcn_wmma_f32_16x16x32_bf16(false, a0.v, false, b1.v,
                                                          (short)0, acc[0][1], false, false);
      acc[1][0] = __builtin_amdgcn_wmma_f32_16x16x32_bf16(false, a1.v, false, b0.v,
                                                          (short)0, acc[1][0], false, false);
      acc[1][1] = __builtin_amdgcn_wmma_f32_16x16x32_bf16(false, a1.v, false, b1.v,
                                                          (short)0, acc[1][1], false, false);
    }
    __syncthreads();
  }

  // epilogue: scale by demod factor, store fp32
  const float* dr = dmod + b * C_OUT;
  float* ob = out + (size_t)b * C_OUT * NPIX + (size_t)pxBlk * 128;
  for (int mi = 0; mi < 2; ++mi)
    for (int ni = 0; ni < 2; ++ni) {
      int p = waveN * 32 + ni * 16 + l15;
      for (int r = 0; r < 8; ++r) {
        int co = coBlk * 64 + waveM * 32 + mi * 16 + r + 8 * half;
        ob[(size_t)co * NPIX + p] = acc[mi][ni][r] * dr[co];
      }
    }
}

extern "C" void kernel_launch(void* const* d_in, const int* in_sizes, int n_in,
                              void* d_out, int out_size, void* d_ws, size_t ws_size,
                              hipStream_t stream) {
  const float* x    = (const float*)d_in[0];
  const float* y    = (const float*)d_in[1];
  const float* wgt  = (const float*)d_in[2];
  const float* mw   = (const float*)d_in[3];
  const float* bias = (const float*)d_in[4];
  float* out = (float*)d_out;

  char* ws = (char*)d_ws;
  float*  s   = (float*)ws;  ws += 8 * 512 * 4;        // 16 KB
  float*  wsq = (float*)ws;  ws += 512 * 512 * 4;      // 1 MB
  float*  dm  = (float*)ws;  ws += 8 * 512 * 4;        // 16 KB
  __bf16* Wp  = (__bf16*)ws; ws += 9 * 512 * 512 * 2;  // 4.7 MB
  __bf16* Xs  = (__bf16*)ws;                           // 9.5 MB

  style_k<<<16,    256, 0, stream>>>(y, mw, bias, s);
  wsq_k  <<<1024,  256, 0, stream>>>(wgt, wsq);
  demod_k<<<16,    256, 0, stream>>>(s, wsq, dm);
  packw_k<<<9216,  256, 0, stream>>>(wgt, Wp);
  packx_k<<<18496, 256, 0, stream>>>(x, s, Xs);

  dim3 grid(8, 8, 8);   // pxBlk, coBlk, batch
  modconv_wmma<<<grid, 256, 0, stream>>>(Xs, Wp, dm, out);
}
